// MHA_61366492725706
// MI455X (gfx1250) — matrix operand order repaired and checked by reading, not compiled, not run on hardware
//
#include <hip/hip_runtime.h>
#include <hip/hip_bf16.h>

typedef __bf16 bf16_t;
typedef __attribute__((ext_vector_type(16))) __bf16 bf16x16;
typedef __attribute__((ext_vector_type(8)))  __bf16 bf16x8;
typedef __attribute__((ext_vector_type(8)))  float  f32x8;

#define WMMA_BF16(A, B, C) \
    __builtin_amdgcn_wmma_f32_16x16x32_bf16(false, (A), false, (B), (short)0, (C), false, false)

static constexpr int B_   = 2;
static constexpr int S_   = 2048;
static constexpr int E_   = 2048;
static constexpr int H_   = 32;
static constexpr int HKV_ = 8;
static constexpr int D_   = 64;
static constexpr int QKV_OUT = (H_ + 2 * HKV_) * D_;  // 3072
static constexpr int M_   = B_ * S_;                  // 4096
static constexpr float SCALE  = 0.125f;               // rsqrt(64)
static constexpr float NEGV   = -10000.0f;
static constexpr float LOG2E  = 1.4426950408889634f;

static __device__ __forceinline__ bf16x16 combine16(bf16x8 lo, bf16x8 hi) {
    return __builtin_shufflevector(lo, hi, 0,1,2,3,4,5,6,7,8,9,10,11,12,13,14,15);
}

// ---------------- fp32 -> bf16 convert (4 elems / thread) ----------------
__global__ void k_cvt_bf16(const float* __restrict__ in, bf16_t* __restrict__ out, int n) {
    int i = (blockIdx.x * blockDim.x + threadIdx.x) * 4;
    if (i + 3 < n) {
        float4 v = *reinterpret_cast<const float4*>(in + i);
        out[i + 0] = (bf16_t)v.x;
        out[i + 1] = (bf16_t)v.y;
        out[i + 2] = (bf16_t)v.z;
        out[i + 3] = (bf16_t)v.w;
    }
}

// -------- fp32 [rows,cols] -> bf16 transposed [cols,rows], 32x32 LDS tile --------
__global__ void k_transpose_bf16(const float* __restrict__ in, bf16_t* __restrict__ out,
                                 int rows, int cols) {
    __shared__ float tile[32][33];
    const int n0 = blockIdx.x * 32;   // col base in input
    const int k0 = blockIdx.y * 32;   // row base in input
    for (int i = threadIdx.y; i < 32; i += 8)
        tile[i][threadIdx.x] = in[(size_t)(k0 + i) * cols + n0 + threadIdx.x];
    __syncthreads();
    for (int i = threadIdx.y; i < 32; i += 8)
        out[(size_t)(n0 + i) * rows + k0 + threadIdx.x] = (bf16_t)tile[threadIdx.x][i];
}

// ---------------- bf16 WMMA GEMM: C[M,N] = A[M,K] * Bt[N,K]^T + bias ----------------
// block = 8 waves (4 m x 2 n); each wave computes a 64x64 C tile (16 WMMAs / k-step).
// Block tile 256(M) x 128(N); K-step 32. ~32 FLOP per L2 byte per wave.
__global__ __launch_bounds__(256) void k_gemm_bf16(
        const bf16_t* __restrict__ A, const bf16_t* __restrict__ Bt,
        const float* __restrict__ bias, float* __restrict__ C,
        int Mdim, int Ndim, int Kdim) {
    const int lane = threadIdx.x & 31;
    const int wave = threadIdx.x >> 5;
    const int wm = wave >> 1, wn = wave & 1;
    const int mBase = blockIdx.y * 256 + wm * 64;
    const int nBase = blockIdx.x * 128 + wn * 64;
    const int half = lane >> 4, nl = lane & 15;
    const int ak = half * 8;     // A-layout K base within 32-chunk
    const int bk = half * 8;     // B-layout K base within 32-chunk

    f32x8 acc[4][4] = {};
    for (int kk = 0; kk < Kdim; kk += 32) {
        bf16x16 af[4], bfm[4];
#pragma unroll
        for (int i = 0; i < 4; i++) {
            const bf16_t* p = A + (size_t)(mBase + i * 16 + nl) * Kdim + kk + ak;
            bf16x8 lo = *reinterpret_cast<const bf16x8*>(p);
            bf16x8 hi = *reinterpret_cast<const bf16x8*>(p + 16);
            af[i] = combine16(lo, hi);
        }
#pragma unroll
        for (int j = 0; j < 4; j++) {
            const bf16_t* pb = Bt + (size_t)(nBase + j * 16 + nl) * Kdim + kk + bk;
            bfm[j] = combine16(*reinterpret_cast<const bf16x8*>(pb),
                               *reinterpret_cast<const bf16x8*>(pb + 16));
        }
#pragma unroll
        for (int i = 0; i < 4; i++)
#pragma unroll
            for (int j = 0; j < 4; j++)
                acc[i][j] = WMMA_BF16(af[i], bfm[j], acc[i][j]);
    }

    float bv[4];
#pragma unroll
    for (int j = 0; j < 4; j++) bv[j] = bias[nBase + j * 16 + nl];
#pragma unroll
    for (int i = 0; i < 4; i++)
#pragma unroll
        for (int j = 0; j < 4; j++) {
            const int n = nBase + j * 16 + nl;
#pragma unroll
            for (int r = 0; r < 8; r++) {
                const int m = mBase + i * 16 + half * 8 + r;
                C[(size_t)m * Ndim + n] = acc[i][j][r] + bv[j];
            }
        }
}

// ------------- rotary + split qkv fp32 -> q/k bf16 [b,h,s,d], v^T bf16 [b,h,d,s] -------------
// 1 warp per (b,s,head): head 0..31 = q, 32..39 = k, 40..47 = v
__global__ __launch_bounds__(256) void k_rope_split(
        const float* __restrict__ qkv, const float* __restrict__ cosT,
        const float* __restrict__ sinT, bf16_t* __restrict__ q,
        bf16_t* __restrict__ k, bf16_t* __restrict__ vT) {
    const int hg = blockIdx.x * 8 + (threadIdx.x >> 5);
    const int lane = threadIdx.x & 31;
    const int b = hg / (S_ * 48);
    const int rem = hg % (S_ * 48);
    const int s = rem / 48;
    const int head = rem % 48;
    const int off = head < 32 ? head * 64
                  : head < 40 ? 2048 + (head - 32) * 64
                              : 2560 + (head - 40) * 64;
    const float* row = qkv + (size_t)(b * S_ + s) * QKV_OUT + off;
    int d1, d2; float o1, o2;
    if (lane < 16) {
        const float x1 = row[lane], x2 = row[lane + 16];
        if (head < 40) {  // rotary on q and k only
            const float c = cosT[s * 16 + lane], sn = sinT[s * 16 + lane];
            o1 = x1 * c - x2 * sn;
            o2 = x1 * sn + x2 * c;
        } else { o1 = x1; o2 = x2; }
        d1 = lane; d2 = lane + 16;
    } else {
        const int j = lane - 16;
        d1 = 32 + j; d2 = 48 + j;
        o1 = row[d1]; o2 = row[d2];
    }
    if (head < 32) {
        bf16_t* dst = q + ((size_t)(b * H_ + head) * S_ + s) * D_;
        dst[d1] = (bf16_t)o1; dst[d2] = (bf16_t)o2;
    } else if (head < 40) {
        bf16_t* dst = k + ((size_t)(b * HKV_ + head - 32) * S_ + s) * D_;
        dst[d1] = (bf16_t)o1; dst[d2] = (bf16_t)o2;
    } else {
        bf16_t* dst = vT + (size_t)(b * HKV_ + head - 40) * D_ * S_;
        dst[(size_t)d1 * S_ + s] = (bf16_t)o1;
        dst[(size_t)d2 * S_ + s] = (bf16_t)o2;
    }
}

// ---------------- flash attention: 4 waves, 16 query rows each, 64-key tiles ----------------
__global__ __launch_bounds__(128) void k_attention(
        const bf16_t* __restrict__ q, const bf16_t* __restrict__ kmat,
        const bf16_t* __restrict__ vT, bf16_t* __restrict__ attn) {
    __shared__ float  sScores[4][16][64];
    __shared__ __align__(32) bf16_t sP[4][16][64];
    __shared__ float  sAlpha[4][16];

    const int lane = threadIdx.x & 31;
    const int wave = threadIdx.x >> 5;
    const int half = lane >> 4, nl = lane & 15;
    const int q0 = blockIdx.x * 64;
    const int b = blockIdx.y / H_, h = blockIdx.y % H_;
    const int hkv = h / (H_ / HKV_);
    const bf16_t* qbase = q    + (size_t)(b * H_   + h)   * S_ * D_;
    const bf16_t* kbase = kmat + (size_t)(b * HKV_ + hkv) * S_ * D_;
    const bf16_t* vbase = vT   + (size_t)(b * HKV_ + hkv) * D_ * S_;
    const int rowBase = q0 + wave * 16;

    // Q fragments (A-layout), held in registers for the whole kernel
    bf16x16 qf[2];
#pragma unroll
    for (int ks = 0; ks < 2; ks++) {
        const bf16_t* p = qbase + (size_t)(rowBase + nl) * D_ + ks * 32 + half * 8;
        qf[ks] = combine16(*reinterpret_cast<const bf16x8*>(p),
                           *reinterpret_cast<const bf16x8*>(p + 16));
    }

    f32x8 oacc[4] = {};
    // lanes L and L+16 both track row (L & 15): identical redundant state
    float m_i = -3.0e38f, l_i = 0.0f;
    const int nkt = q0 / 64 + 1;  // causal: only key tiles <= diagonal

    for (int kt = 0; kt < nkt; kt++) {
        // ---- scores S = (Q K^T) * scale, masked, staged to LDS ----
#pragma unroll
        for (int nt = 0; nt < 4; nt++) {
            f32x8 sc = {};
#pragma unroll
            for (int ks = 0; ks < 2; ks++) {
                const bf16_t* p = kbase + (size_t)(kt * 64 + nt * 16 + nl) * D_
                                + ks * 32 + half * 8;
                bf16x16 bfk = combine16(*reinterpret_cast<const bf16x8*>(p),
                                        *reinterpret_cast<const bf16x8*>(p + 16));
                sc = WMMA_BF16(qf[ks], bfk, sc);
            }
            const int gkey = kt * 64 + nt * 16 + nl;
#pragma unroll
            for (int r = 0; r < 8; r++) {
                const int mrow = r + half * 8;
                float v = sc[r] * SCALE;
                if (gkey > rowBase + mrow) v = NEGV;
                sScores[wave][mrow][nt * 16 + nl] = v;
            }
        }
        __syncthreads();

        // ---- online softmax stats: row (lane&15), columns split across lane halves ----
        {
            const int c0 = half * 32;  // this half-lane scans 32 of the 64 columns
            float tmax = -3.0e38f;
            for (int j = 0; j < 32; j++)
                tmax = fmaxf(tmax, sScores[wave][nl][c0 + j]);
            tmax = fmaxf(tmax, __shfl_xor(tmax, 16, 32));
            const float mnew = fmaxf(m_i, tmax);
            const float al   = exp2f((m_i - mnew) * LOG2E);
            float sum = 0.0f;
            for (int j = 0; j < 32; j++) {
                const float p = exp2f((sScores[wave][nl][c0 + j] - mnew) * LOG2E);
                sP[wave][nl][c0 + j] = (bf16_t)p;
                sum += p;
            }
            sum += __shfl_xor(sum, 16, 32);
            l_i = l_i * al + sum;
            m_i = mnew;
            if (lane < 16) sAlpha[wave][nl] = al;
        }
        __syncthreads();

        // ---- rescale O accumulators ----
        float aRow[8];
#pragma unroll
        for (int r = 0; r < 8; r++) aRow[r] = sAlpha[wave][r + half * 8];
#pragma unroll
        for (int nt = 0; nt < 4; nt++)
#pragma unroll
            for (int r = 0; r < 8; r++) oacc[nt][r] *= aRow[r];

        // ---- O += P * V  (P from LDS in A-layout, V^T from global in B-layout) ----
#pragma unroll
        for (int ks = 0; ks < 2; ks++) {
            const bf16_t* pp = &sP[wave][nl][ks * 32 + half * 8];
            bf16x16 af = combine16(*reinterpret_cast<const bf16x8*>(pp),
                                   *reinterpret_cast<const bf16x8*>(pp + 16));
#pragma unroll
            for (int nt = 0; nt < 4; nt++) {
                const bf16_t* vp = vbase + (size_t)(nt * 16 + nl) * S_
                                 + kt * 64 + ks * 32 + half * 8;
                bf16x16 bfv = combine16(*reinterpret_cast<const bf16x8*>(vp),
                                        *reinterpret_cast<const bf16x8*>(vp + 16));
                oacc[nt] = WMMA_BF16(af, bfv, oacc[nt]);
            }
        }
        __syncthreads();
    }

    // ---- normalize + store bf16 [b, s, h*64+d] ----
    if (lane < 16) sAlpha[wave][nl] = 1.0f / l_i;
    __syncthreads();
    float inv[8];
#pragma unroll
    for (int r = 0; r < 8; r++) inv[r] = sAlpha[wave][r + half * 8];
#pragma unroll
    for (int nt = 0; nt < 4; nt++)
#pragma unroll
        for (int r = 0; r < 8; r++) {
            const int grow = rowBase + r + half * 8;
            const int col  = h * 64 + nt * 16 + nl;
            attn[(size_t)(b * S_ + grow) * (H_ * D_) + col] = (bf16_t)(oacc[nt][r] * inv[r]);
        }
}

extern "C" void kernel_launch(void* const* d_in, const int* in_sizes, int n_in,
                              void* d_out, int out_size, void* d_ws, size_t ws_size,
                              hipStream_t stream) {
    const float* x    = (const float*)d_in[0];
    const float* Wqkv = (const float*)d_in[1];
    const float* bqkv = (const float*)d_in[2];
    const float* Wout = (const float*)d_in[3];
    const float* bout = (const float*)d_in[4];
    const float* cosT = (const float*)d_in[5];
    const float* sinT = (const float*)d_in[6];

    char* ws = (char*)d_ws;
    size_t off = 0;
    bf16_t* xbf   = (bf16_t*)(ws + off); off += (size_t)M_ * E_ * 2;          // 16 MB
    bf16_t* wqkvT = (bf16_t*)(ws + off); off += (size_t)QKV_OUT * E_ * 2;     // 12 MB
    bf16_t* woutT = (bf16_t*)(ws + off); off += (size_t)E_ * (H_ * D_) * 2;   // 8 MB
    float*  qkv   = (float*) (ws + off); off += (size_t)M_ * QKV_OUT * 4;     // 48 MB
    bf16_t* qb    = (bf16_t*)(ws + off); off += (size_t)B_ * H_   * S_ * D_ * 2;
    bf16_t* kb    = (bf16_t*)(ws + off); off += (size_t)B_ * HKV_ * S_ * D_ * 2;
    bf16_t* vTb   = (bf16_t*)(ws + off); off += (size_t)B_ * HKV_ * D_ * S_ * 2;
    bf16_t* attn  = (bf16_t*)(ws + off); off += (size_t)M_ * H_ * D_ * 2;

    // 1) x -> bf16
    k_cvt_bf16<<<(M_ * E_) / 1024, 256, 0, stream>>>(x, xbf, M_ * E_);
    // 2) weights -> bf16 transposed (contiguous-K B-fragment loads)
    k_transpose_bf16<<<dim3(QKV_OUT / 32, E_ / 32), dim3(32, 8), 0, stream>>>(
        Wqkv, wqkvT, E_, QKV_OUT);
    k_transpose_bf16<<<dim3(E_ / 32, (H_ * D_) / 32), dim3(32, 8), 0, stream>>>(
        Wout, woutT, H_ * D_, E_);
    // 3) QKV projection (bias fused): block tile 256x128
    k_gemm_bf16<<<dim3(QKV_OUT / 128, M_ / 256), 256, 0, stream>>>(
        xbf, wqkvT, bqkv, qkv, M_, QKV_OUT, E_);
    // 4) rotary + split to q/k and transposed v
    k_rope_split<<<(B_ * S_ * 48) / 8, 256, 0, stream>>>(qkv, cosT, sinT, qb, kb, vTb);
    // 5) causal flash attention
    k_attention<<<dim3(S_ / 64, B_ * H_), 128, 0, stream>>>(qb, kb, vTb, attn);
    // 6) output projection (bias fused) -> fp32 d_out
    k_gemm_bf16<<<dim3(E_ / 128, M_ / 256), 256, 0, stream>>>(
        attn, woutT, bout, (float*)d_out, M_, E_, H_ * D_);
}
